// StackLSTM_17806934409741
// MI455X (gfx1250) — matrix operand order repaired and checked by reading, not compiled, not run on hardware
//
#include <hip/hip_runtime.h>
#include <hip/hip_bf16.h>

// ---------------------------------------------------------------------------
// StackLSTM (T=256, B=64, H=1024, L=2) for gfx1250 (MI455X, wave32, WMMA).
//
//  * gates = [x|h] @ [W_ih|W_hh]^T, bf16 WMMA (v_wmma_f32_16x16x32_bf16).
//  * Weights fragment-packed to bf16 (ISA 7.12.2 B layout), L2-resident.
//  * input_seq pre-converted to bf16 once; h state kept in bf16; c in f32.
//  * Split-K x2: even wave of a pair does x@W_ih (k 0..31), odd wave does
//    h@W_hh (k 32..63); partial sums combined via LDS, even wave does the
//    fused i/f/g/o sigmoid/tanh + cell-update epilogue from registers.
//  * Software-pipelined fragment loads (double buffer) hide L2 latency.
//  * 512 sequential kernel launches implement the recurrence; h ping-pongs.
// ---------------------------------------------------------------------------

typedef __attribute__((ext_vector_type(16))) __bf16 v16bf;
typedef __attribute__((ext_vector_type(8)))  float  v8f;

#define HSZ   1024u
#define TSTEPS 256
#define GATES4H 4096u                              // 4*H
#define ELEMS_PER_LAYER (GATES4H * 2u * HSZ)       // 8388608 packed bf16 / layer
#define TILECOL_ELEMS   (64u * 32u * 16u)          // 32768 elems per n-tile col
#define GSTRIDE         (64u * TILECOL_ELEMS)      // elems per gate block (4 MB)
#define XSEQ_ELEMS      (256u * 64u * 1024u)       // 16777216
#define KLOC  32u                                  // k-steps per wave (split-K 2)

static __device__ __forceinline__ unsigned short f2bf(float f) {
    // round-to-nearest-even f32 -> bf16
    unsigned int u = __float_as_uint(f);
    unsigned int r = u + 0x7FFFu + ((u >> 16) & 1u);
    return (unsigned short)(r >> 16);
}

static __device__ __forceinline__ float sigmoidf_(float x) {
    return 1.0f / (1.0f + __expf(-x));
}

union Frag { v16bf v; uint4 q[2]; };
union V8u  { v8f v; float4 q[2]; };

// ---------------------------------------------------------------------------
// Pack [W_ih | W_hh] (f32, row-major [L,4H,H]) into bf16 B-fragment layout:
//   elem index = ((gt*64 + kstep)*32 + lane)*16 + e
//   lane: col = lane&15 -> N, half = lane>>4; klocal = (e&8)*2 + half*8 + (e&7)
// ---------------------------------------------------------------------------
__global__ __launch_bounds__(256) void pack_weights_kernel(
        const float* __restrict__ W_ih, const float* __restrict__ W_hh,
        unsigned short* __restrict__ wpack) {
    const unsigned idx = blockIdx.x * blockDim.x + threadIdx.x; // 0..2*8388608-1
    const unsigned l   = idx / ELEMS_PER_LAYER;
    const unsigned rem = idx % ELEMS_PER_LAYER;
    const unsigned e    = rem & 15u;
    const unsigned lane = (rem >> 4) & 31u;
    const unsigned k    = (rem >> 9) & 63u;
    const unsigned gt   = rem >> 15;                 // 0..255 (global n-tile)
    const unsigned n    = gt * 16u + (lane & 15u);   // gate-row in [0,4096)
    const unsigned K    = k * 32u + ((e & 8u) << 1) + ((lane >> 4) << 3) + (e & 7u);
    float v;
    if (K < HSZ) v = W_ih[(size_t)l * GATES4H * HSZ + (size_t)n * HSZ + K];
    else         v = W_hh[(size_t)l * GATES4H * HSZ + (size_t)n * HSZ + (K - HSZ)];
    wpack[(size_t)l * ELEMS_PER_LAYER + rem] = f2bf(v);
}

// Convert the whole input sequence to bf16 once (each element exactly once).
__global__ __launch_bounds__(256) void cvt_x_kernel(
        const float* __restrict__ xin, unsigned short* __restrict__ xbf) {
    const unsigned i = blockIdx.x * blockDim.x + threadIdx.x;
    xbf[i] = f2bf(xin[i]);
}

// h0 -> bf16 ping buffers, c0 -> f32 state.
__global__ __launch_bounds__(256) void init_state_kernel(
        const float* __restrict__ h0, const float* __restrict__ c0,
        unsigned short* __restrict__ hbf, float* __restrict__ cbuf) {
    const unsigned i = blockIdx.x * blockDim.x + threadIdx.x; // 0..131071
    const unsigned l = i >> 16;          // layer
    const unsigned r = i & 65535u;       // within [B*H]
    hbf[((size_t)l * 2u + 0u) * 65536u + r] = f2bf(h0[i]);   // ping buffer 0
    cbuf[(size_t)l * 65536u + r]            = c0[i];
}

// ---------------------------------------------------------------------------
// One LSTM cell step. grid = 64 blocks x 256 threads = 512 waves.
// Wave pair -> one (m_tile, nt) tile; even wave K=[0,1024) over x, odd wave
// K=[1024,2048) over h. Pipelined b128 loads feed 4 WMMAs per k-step.
// ---------------------------------------------------------------------------
__global__ __launch_bounds__(256) void lstm_step_kernel(
        const unsigned short* __restrict__ x,     // [64,1024] bf16 layer input
        const unsigned short* __restrict__ h_in,  // [64,1024] bf16
        unsigned short* __restrict__ h_out,       // [64,1024] bf16
        float* __restrict__ c,                    // [64,1024] f32 (in place)
        const unsigned short* __restrict__ wpack, // this layer's packed W
        const float* __restrict__ b_ih,           // [4096]
        const float* __restrict__ b_hh,           // [4096]
        float* __restrict__ final_out)            // nullptr or [64,1024] f32
{
    __shared__ float red[4][4][32 * 8];           // [pair_local][gate][lane*8+r]

    const unsigned lane   = threadIdx.x & 31u;
    const unsigned wid_l  = threadIdx.x >> 5;           // 0..7
    const unsigned wid_g  = blockIdx.x * 8u + wid_l;    // 0..511
    const unsigned kk     = wid_g & 1u;                 // K half: 0 -> x, 1 -> h
    const unsigned pair   = wid_g >> 1;                 // 0..255 tile id
    const unsigned m_tile = pair & 3u;                  // batch tile (B=64 -> 4)
    const unsigned nt     = pair >> 2;                  // 0..63 per gate block
    const unsigned pl     = wid_l >> 1;                 // pair within block 0..3
    const unsigned col    = lane & 15u;
    const unsigned half   = lane >> 4;

    // A-fragment (ISA 7.12.2, 16-bit A 16x32): lane&15 -> M; lane>>4 selects
    // K runs [8h, 8h+8) and [16+8h, 16+8h+8) within each 32-wide k-step.
    const unsigned arow = m_tile * 16u + col;
    const unsigned short* asrc = (kk == 0u) ? x : h_in;     // fixed per wave
    const unsigned short* ap0  = asrc + (size_t)arow * HSZ + half * 8u;
    const unsigned short* bp0  = wpack + (size_t)nt * TILECOL_ELEMS
                               + (size_t)(kk * KLOC) * 512u + (size_t)lane * 16u;

    v8f acc[4] = {{}, {}, {}, {}};                      // i, f, g, o

    // Software-pipelined K loop: issue (ks+1) loads before (ks) WMMAs.
    Frag a[2];
    Frag b[2][4];
    a[0].q[0] = *(const uint4*)(ap0);
    a[0].q[1] = *(const uint4*)(ap0 + 16);
    #pragma unroll
    for (unsigned g = 0; g < 4u; ++g) {
        b[0][g].q[0] = *(const uint4*)(bp0 + (size_t)g * GSTRIDE);
        b[0][g].q[1] = *(const uint4*)(bp0 + (size_t)g * GSTRIDE + 8);
    }

    #pragma unroll 2
    for (unsigned ks = 0; ks < KLOC; ++ks) {
        const unsigned cur = ks & 1u;
        if (ks + 1u < KLOC) {
            const unsigned nxt = cur ^ 1u;
            const unsigned short* apn = ap0 + (size_t)(ks + 1u) * 32u;
            const unsigned short* bpn = bp0 + (size_t)(ks + 1u) * 512u;
            a[nxt].q[0] = *(const uint4*)(apn);
            a[nxt].q[1] = *(const uint4*)(apn + 16);
            #pragma unroll
            for (unsigned g = 0; g < 4u; ++g) {
                b[nxt][g].q[0] = *(const uint4*)(bpn + (size_t)g * GSTRIDE);
                b[nxt][g].q[1] = *(const uint4*)(bpn + (size_t)g * GSTRIDE + 8);
            }
        }
        acc[0] = __builtin_amdgcn_wmma_f32_16x16x32_bf16(false, a[cur].v, false,
                     b[cur][0].v, (short)0, acc[0], false, false);
        acc[1] = __builtin_amdgcn_wmma_f32_16x16x32_bf16(false, a[cur].v, false,
                     b[cur][1].v, (short)0, acc[1], false, false);
        acc[2] = __builtin_amdgcn_wmma_f32_16x16x32_bf16(false, a[cur].v, false,
                     b[cur][2].v, (short)0, acc[2], false, false);
        acc[3] = __builtin_amdgcn_wmma_f32_16x16x32_bf16(false, a[cur].v, false,
                     b[cur][3].v, (short)0, acc[3], false, false);
    }

    // Split-K reduction: odd wave parks its partials in LDS, even wave sums.
    if (kk == 1u) {
        #pragma unroll
        for (unsigned g = 0; g < 4u; ++g) {
            V8u t; t.v = acc[g];
            float* dst = &red[pl][g][lane * 8u];
            *(float4*)(dst)      = t.q[0];
            *(float4*)(dst + 4u) = t.q[1];
        }
    }
    __syncthreads();
    if (kk == 1u) return;

    #pragma unroll
    for (unsigned g = 0; g < 4u; ++g) {
        const float* s = &red[pl][g][lane * 8u];
        const float4 lo = *(const float4*)(s);
        const float4 hi = *(const float4*)(s + 4u);
        acc[g][0] += lo.x; acc[g][1] += lo.y; acc[g][2] += lo.z; acc[g][3] += lo.w;
        acc[g][4] += hi.x; acc[g][5] += hi.y; acc[g][6] += hi.z; acc[g][7] += hi.w;
    }

    // Epilogue: bias + nonlinearity + cell update, straight from accumulators.
    const unsigned ncol = nt * 16u + col;                    // 0..1023
    const float bi = b_ih[ncol]           + b_hh[ncol];
    const float bf = b_ih[HSZ + ncol]     + b_hh[HSZ + ncol];
    const float bg = b_ih[2u*HSZ + ncol]  + b_hh[2u*HSZ + ncol];
    const float bo = b_ih[3u*HSZ + ncol]  + b_hh[3u*HSZ + ncol];

    #pragma unroll
    for (unsigned r = 0; r < 8u; ++r) {
        const unsigned row = m_tile * 16u + half * 8u + r;   // C/D: V r, lane half
        const size_t   idx = (size_t)row * HSZ + ncol;
        const float iv = sigmoidf_(acc[0][r] + bi);
        const float fv = sigmoidf_(acc[1][r] + bf);
        const float gv = tanhf(acc[2][r] + bg);
        const float ov = sigmoidf_(acc[3][r] + bo);
        const float cn = fv * c[idx] + iv * gv;
        c[idx] = cn;
        const float hn = ov * tanhf(cn);
        h_out[idx] = f2bf(hn);               // bf16 state for next GEMMs
        if (final_out) final_out[idx] = hn;  // f32 final output at t = T-1
    }
}

// ---------------------------------------------------------------------------
// Host: pack weights, convert x, init state, then T x L sequential steps.
// ---------------------------------------------------------------------------
extern "C" void kernel_launch(void* const* d_in, const int* in_sizes, int n_in,
                              void* d_out, int out_size, void* d_ws, size_t ws_size,
                              hipStream_t stream) {
    const float* input_seq = (const float*)d_in[0]; // [256,64,1024]
    const float* h0        = (const float*)d_in[1]; // [2,64,1024]
    const float* c0        = (const float*)d_in[2]; // [2,64,1024]
    const float* W_ih      = (const float*)d_in[3]; // [2,4096,1024]
    const float* W_hh      = (const float*)d_in[4]; // [2,4096,1024]
    const float* b_ih      = (const float*)d_in[5]; // [2,4096]
    const float* b_hh      = (const float*)d_in[6]; // [2,4096]
    float* out = (float*)d_out;                     // [64,1024]

    char* ws = (char*)d_ws;
    unsigned short* wpack = (unsigned short*)ws;                        // 32 MB
    unsigned short* xbf   = (unsigned short*)(ws + (size_t)2 * ELEMS_PER_LAYER * 2u); // 32 MB
    unsigned short* hbf   = xbf + (size_t)XSEQ_ELEMS;                   // [L][2][65536] bf16
    float*          cbuf  = (float*)(hbf + (size_t)4 * 65536u);         // [L][65536] f32

    // 1) bf16 fragment-pack the weights (deterministic, every call)
    pack_weights_kernel<<<(2u * ELEMS_PER_LAYER) / 256u, 256, 0, stream>>>(
        W_ih, W_hh, wpack);

    // 2) convert input sequence to bf16 (each element exactly once)
    cvt_x_kernel<<<XSEQ_ELEMS / 256u, 256, 0, stream>>>(input_seq, xbf);

    // 3) initialize h/c state from inputs
    init_state_kernel<<<(2u * 65536u) / 256u, 256, 0, stream>>>(h0, c0, hbf, cbuf);

    // 4) sequential recurrence: 256 steps x 2 layers
    for (int t = 0; t < TSTEPS; ++t) {
        const int pi = t & 1, po = pi ^ 1;
        // layer 0: x = input_seq[t] (bf16)
        lstm_step_kernel<<<64, 256, 0, stream>>>(
            xbf + (size_t)t * 65536u,
            hbf + (size_t)(0 * 2 + pi) * 65536u,
            hbf + (size_t)(0 * 2 + po) * 65536u,
            cbuf + (size_t)0 * 65536u,
            wpack + (size_t)0 * ELEMS_PER_LAYER,
            b_ih + 0, b_hh + 0,
            nullptr);
        // layer 1: x = layer-0 h at this timestep (bf16)
        lstm_step_kernel<<<64, 256, 0, stream>>>(
            hbf + (size_t)(0 * 2 + po) * 65536u,
            hbf + (size_t)(1 * 2 + pi) * 65536u,
            hbf + (size_t)(1 * 2 + po) * 65536u,
            cbuf + (size_t)1 * 65536u,
            wpack + (size_t)1 * ELEMS_PER_LAYER,
            b_ih + GATES4H, b_hh + GATES4H,
            (t == TSTEPS - 1) ? out : nullptr);
    }
}
